// ConvLayer_76879914598811
// MI455X (gfx1250) — matrix operand order repaired and checked by reading, not compiled, not run on hardware
//
#include <hip/hip_runtime.h>
#include <hip/hip_bf16.h>
#include <math.h>

typedef __attribute__((ext_vector_type(16))) _Float16 v16h;
typedef __attribute__((ext_vector_type(8)))  _Float16 v8h;
typedef __attribute__((ext_vector_type(8)))  float    v8f;

#define NN 20000
#define MM 32
#define EE 20
#define HH 128

// ---------------- workspace (f16) layout, in halves ----------------
#define WT_ELEMS   (256 * 384)     // Wfull^T per branch (98304)
#define WFT_ELEMS  (128 * 32)      // Wf^T (K padded 20->32) per branch (4096)
#define WS_WT(br)  ((size_t)(br) * WT_ELEMS)
#define WS_WFT(br) ((size_t)(2 * WT_ELEMS) + (size_t)(br) * WFT_ELEMS)

// ---------------- dynamic LDS layout, in bytes ----------------
#define BT_STRIDE   392                          // halves per Wfull^T row (384 + 8 pad)
#define LDS_BT      0
#define LDS_BT_SZ   (256 * BT_STRIDE * 2)        // 200704
#define WFT_STRIDE  40                           // halves per Wf^T row (32 + 8 pad)
#define LDS_WFT     (LDS_BT + LDS_BT_SZ)
#define LDS_WFT_SZ  (128 * WFT_STRIDE * 2)       // 10240
#define EDGE_STRIDE 136                          // halves per staged edge row (128 + 8 pad)
#define LDS_EDGE    (LDS_WFT + LDS_WFT_SZ)
#define LDS_EDGE_SZ (8 * 16 * EDGE_STRIDE * 2)   // 34816
#define LDS_ACC     (LDS_EDGE + LDS_EDGE_SZ)
#define LDS_ACC_SZ  (4 * 128 * 4)                // 2048
#define LDS_TOTAL   (LDS_ACC + LDS_ACC_SZ)       // 247808 B < 320 KB WGP LDS

#define WMMA_F16(A, B, C) \
  __builtin_amdgcn_wmma_f32_16x16x32_f16(false, (A), false, (B), (short)0, (C), false, false)

__device__ __forceinline__ void cvt4(v16h& a, int o, float4 x) {
  a[o + 0] = (_Float16)x.x; a[o + 1] = (_Float16)x.y;
  a[o + 2] = (_Float16)x.z; a[o + 3] = (_Float16)x.w;
}

// B fragment: e0..7 at p, e8..15 at p+16 halves (Wfull^T row, offset half*8)
__device__ __forceinline__ v16h lds_b16(const _Float16* p) {
  v8h lo = *(const v8h*)p;
  v8h hi = *(const v8h*)(p + 16);
  return __builtin_shufflevector(lo, hi, 0,1,2,3,4,5,6,7,8,9,10,11,12,13,14,15);
}

// A fragment from staged f16 row: e0..7 at p, e8..15 at p+16 halves (K and K+16)
__device__ __forceinline__ v16h lds_a16(const _Float16* p) {
  v8h lo = *(const v8h*)p;
  v8h hi = *(const v8h*)(p + 16);
  return __builtin_shufflevector(lo, hi, 0,1,2,3,4,5,6,7,8,9,10,11,12,13,14,15);
}

// A fragment built from an f32 feature row: K-chunk kc32 (multiple of 32), half lane group
__device__ __forceinline__ v16h make_a_f32(const float* rowp, int kc32, int halfsel) {
  const float* p0 = rowp + kc32 + halfsel * 8;        // K = base .. base+7
  const float* p1 = rowp + kc32 + 16 + halfsel * 8;   // K = base+16 .. base+23
  float4 a0 = *(const float4*)(p0);
  float4 a1 = *(const float4*)(p0 + 4);
  float4 b0 = *(const float4*)(p1);
  float4 b1 = *(const float4*)(p1 + 4);
  v16h r;
  cvt4(r, 0, a0); cvt4(r, 4, a1); cvt4(r, 8, b0); cvt4(r, 12, b1);
  return r;
}

// Async global -> LDS copy of `n16` 16-byte chunks (per-lane addresses).
// Uses the CDNA5 ASYNCcnt path: no VGPR bounce, waits must precede the barrier.
__device__ __forceinline__ void async_copy_to_lds(_Float16* ldsDst, const _Float16* gSrc,
                                                  int n16) {
  unsigned int ldsoff = (unsigned int)(uintptr_t)ldsDst;   // LDS aperture: addr[31:0]
  unsigned long long gaddr = (unsigned long long)(uintptr_t)gSrc;
  for (int i = 0; i < n16; ++i) {
    asm volatile("global_load_async_to_lds_b128 %0, %1, off"
                 :: "v"(ldsoff), "v"(gaddr) : "memory");
    ldsoff += 16;
    gaddr  += 16;
  }
}

__device__ __forceinline__ void async_wait_all() {
  asm volatile("s_wait_asynccnt 0x0" ::: "memory");
}

__device__ __forceinline__ float softplus_f(float x) {
  return (x > 20.0f) ? x : log1pf(__expf(x));
}
__device__ __forceinline__ float sigmoid_f(float x) {
  return 1.0f / (1.0f + __expf(-x));
}

// ---------------------------------------------------------------------------
// Prep: convert weights to f16, transposed, into workspace.
// ---------------------------------------------------------------------------
__global__ void prep_weights_f16(const float* __restrict__ cheWfull,
                                 const float* __restrict__ vdwWfull,
                                 const float* __restrict__ cheWf,
                                 const float* __restrict__ vdwWf,
                                 _Float16* __restrict__ ws) {
  int id = blockIdx.x * blockDim.x + threadIdx.x;
  if (id < WT_ELEMS) {                 // Wfull^T[c][k] = Wfull[k][c]
    int c = id / 384, k = id % 384;
    ws[WS_WT(0) + id] = (_Float16)cheWfull[k * 256 + c];
    ws[WS_WT(1) + id] = (_Float16)vdwWfull[k * 256 + c];
  }
  int id2 = id - WT_ELEMS;
  if (id2 >= 0 && id2 < WFT_ELEMS) {   // Wf^T[h][e], e padded 20 -> 32 with zeros
    int h = id2 / 32, e = id2 % 32;
    _Float16 c = (e < EE) ? (_Float16)cheWf[e * HH + h] : (_Float16)0.0f;
    _Float16 v = (e < EE) ? (_Float16)vdwWf[e * HH + h] : (_Float16)0.0f;
    ws[WS_WFT(0) + id2] = c;
    ws[WS_WFT(1) + id2] = v;
  }
}

// ---------------------------------------------------------------------------
// Fused conv layer: 4 nodes per workgroup, 8 waves, both branches in-WG.
// ---------------------------------------------------------------------------
__global__ void __launch_bounds__(256, 1)
conv_fused_kernel(const float* __restrict__ nodes,
                  const float* __restrict__ cheRbf, const int* __restrict__ cheIdx,
                  const float* __restrict__ vdwRbf, const int* __restrict__ vdwIdx,
                  const float* __restrict__ cheBf,  const float* __restrict__ cheBfull,
                  const float* __restrict__ vdwBf,  const float* __restrict__ vdwBfull,
                  const _Float16* __restrict__ ws,
                  float* __restrict__ out) {
  extern __shared__ unsigned char smem[];
  _Float16* sBT   = (_Float16*)(smem + LDS_BT);    // Wfull^T, stride BT_STRIDE
  _Float16* sWfT  = (_Float16*)(smem + LDS_WFT);   // Wf^T,    stride WFT_STRIDE
  _Float16* sEdge = (_Float16*)(smem + LDS_EDGE);  // per-wave 16 x EDGE_STRIDE
  float*    sAcc  = (float*)(smem + LDS_ACC);      // 4 nodes x 128 accumulators

  const int tid       = threadIdx.x;
  const int wave      = tid >> 5;
  const int lane      = tid & 31;
  const int rowInTile = lane & 15;   // row within 16-row tile / col within 16-col tile
  const int halfsel   = lane >> 4;   // lane half-group select
  const int nodeBlock = blockIdx.x * 4;

  const int node    = nodeBlock + (wave >> 1);   // constant per wave
  const int nodeLoc = wave >> 1;
  const int m       = (wave & 1) * 16 + rowInTile;

  for (int i = tid; i < 512; i += 256) sAcc[i] = 0.0f;

  _Float16* myEdge = sEdge + (size_t)wave * 16 * EDGE_STRIDE;
  const float* selfp = nodes + (size_t)node * HH;

#pragma unroll 1
  for (int br = 0; br < 2; ++br) {
    const float* rbf   = br ? vdwRbf   : cheRbf;
    const int*   idx   = br ? vdwIdx   : cheIdx;
    const float* bf    = br ? vdwBf    : cheBf;
    const float* bfull = br ? vdwBfull : cheBfull;
    const _Float16* gWT  = ws + WS_WT(br);
    const _Float16* gWfT = ws + WS_WFT(br);

    __syncthreads();   // previous branch done reading LDS weights

    // ---- stage Wfull^T (f16) into LDS via async global->LDS DMA:
    //      thread t owns row t (384 halves = 48 x 16B) ----
    async_copy_to_lds(sBT + (size_t)tid * BT_STRIDE, gWT + (size_t)tid * 384, 48);
    // ---- stage Wf^T (f16, padded): threads <128 own one 32-half row (4 x 16B) ----
    if (tid < 128)
      async_copy_to_lds(sWfT + (size_t)tid * WFT_STRIDE, gWfT + (size_t)tid * 32, 4);
    async_wait_all();          // s_wait_asynccnt 0 before signaling the barrier
    __syncthreads();

    // =========== Edge GEMM: rbf(16x20 pad 32) @ Wf -> staged f16 16x128 ======
    const float* rbfRow = rbf + ((size_t)node * MM + m) * EE;
    v16h a_e = {};
    {
      float4 e0 = *(const float4*)(rbfRow + halfsel * 8);
      float4 e1 = *(const float4*)(rbfRow + halfsel * 8 + 4);
      cvt4(a_e, 0, e0); cvt4(a_e, 4, e1);
      if (halfsel == 0) {           // K16..19 valid, K20..31 are zero padding
        float4 e2 = *(const float4*)(rbfRow + 16);
        cvt4(a_e, 8, e2);
        a_e[12] = (_Float16)0.0f; a_e[13] = (_Float16)0.0f;
        a_e[14] = (_Float16)0.0f; a_e[15] = (_Float16)0.0f;
      }
    }
#pragma unroll 1
    for (int j = 0; j < 8; ++j) {
      const _Float16* bp = sWfT + (size_t)(j * 16 + rowInTile) * WFT_STRIDE + halfsel * 8;
      v16h b_e = lds_b16(bp);
      v8f zc = {};
      v8f c = WMMA_F16(a_e, b_e, zc);
      float bb = bf[j * 16 + rowInTile];
#pragma unroll
      for (int v = 0; v < 8; ++v) {
        int r = v + halfsel * 8;
        myEdge[(size_t)r * EDGE_STRIDE + j * 16 + rowInTile] = (_Float16)(c[v] + bb);
      }
    }
    __syncthreads();

    // =========== Build A fragments: [self(4) | edge(4) | nbr(4)] k-chunks ====
    const int nb = idx[(size_t)node * MM + m];
    const float* nbp = nodes + (size_t)nb * HH;
    __builtin_prefetch((const void*)nbp, 0, 1);     // global_prefetch_b8

    v16h afr[12];
#pragma unroll
    for (int kc = 0; kc < 4; ++kc)
      afr[kc] = make_a_f32(selfp, kc * 32, halfsel);
#pragma unroll
    for (int kc = 0; kc < 4; ++kc)
      afr[4 + kc] = lds_a16(myEdge + (size_t)rowInTile * EDGE_STRIDE + kc * 32 + halfsel * 8);
#pragma unroll
    for (int kc = 0; kc < 4; ++kc)
      afr[8 + kc] = make_a_f32(nbp, kc * 32, halfsel);

    // =========== Main GEMM (K=384) + gated activation + neighbor reduce ======
#pragma unroll 1
    for (int j = 0; j < 8; ++j) {
      v8f cf = {}, cc = {};
      const _Float16* rowF = sBT + (size_t)(j * 16 + rowInTile) * BT_STRIDE + halfsel * 8;
      const _Float16* rowC = sBT + (size_t)(128 + j * 16 + rowInTile) * BT_STRIDE + halfsel * 8;
#pragma unroll
      for (int kc = 0; kc < 12; ++kc) {
        v16h Bf = lds_b16(rowF + kc * 32);
        cf = WMMA_F16(afr[kc], Bf, cf);
        v16h Bc = lds_b16(rowC + kc * 32);
        cc = WMMA_F16(afr[kc], Bc, cc);
      }
      float b1 = bfull[j * 16 + rowInTile];
      float b2 = bfull[128 + j * 16 + rowInTile];
      float s = 0.0f;
#pragma unroll
      for (int v = 0; v < 8; ++v) {
        float fv = cf[v] + b1;
        float cv = cc[v] + b2;
        s += sigmoid_f(fv) * softplus_f(cv);
      }
      atomicAdd(&sAcc[nodeLoc * 128 + j * 16 + rowInTile], s);   // ds_add_f32
    }
  }

  __syncthreads();
  // ---- finalize: out = softplus(nodes + che + vdw) ----
  for (int i = tid; i < 512; i += 256) {
    int nl = i >> 7, h = i & 127;
    size_t g = (size_t)(nodeBlock + nl) * HH + h;
    out[g] = softplus_f(nodes[g] + sAcc[i]);
  }
}

// ---------------------------------------------------------------------------
extern "C" void kernel_launch(void* const* d_in, const int* in_sizes, int n_in,
                              void* d_out, int out_size, void* d_ws, size_t ws_size,
                              hipStream_t stream) {
  const float* nodes    = (const float*)d_in[0];
  const float* cheRbf   = (const float*)d_in[1];
  const int*   cheIdx   = (const int*)  d_in[2];
  const float* vdwRbf   = (const float*)d_in[3];
  const int*   vdwIdx   = (const int*)  d_in[4];
  const float* cheWf    = (const float*)d_in[5];
  const float* cheBf    = (const float*)d_in[6];
  const float* cheWfull = (const float*)d_in[7];
  const float* cheBfull = (const float*)d_in[8];
  const float* vdwWf    = (const float*)d_in[9];
  const float* vdwBf    = (const float*)d_in[10];
  const float* vdwWfull = (const float*)d_in[11];
  const float* vdwBfull = (const float*)d_in[12];
  _Float16* ws  = (_Float16*)d_ws;
  float*    out = (float*)d_out;

  (void)in_sizes; (void)n_in; (void)out_size; (void)ws_size;

  // 1) convert weights to f16 (transposed) in workspace
  int prepElems = WT_ELEMS + WFT_ELEMS;          // 102400
  prep_weights_f16<<<(prepElems + 255) / 256, 256, 0, stream>>>(
      cheWfull, vdwWfull, cheWf, vdwWf, ws);

  // 2) fused conv: 4 nodes per WG, 8 waves, both branches, 242 KB dynamic LDS
  conv_fused_kernel<<<NN / 4, 256, LDS_TOTAL, stream>>>(
      nodes, cheRbf, cheIdx, vdwRbf, vdwIdx,
      cheBf, cheBfull, vdwBf, vdwBfull, ws, out);
}
